// CtcAdapter_19421842112910
// MI455X (gfx1250) — compile-verified, matrix operand order checked by reading, not run on hardware
//
#include <hip/hip_runtime.h>
#include <hip/hip_bf16.h>
#include <math.h>

// Problem sizes (fixed by the reference)
#define B_  16
#define T_  2048
#define D_  1024
#define V_  256
#define DO_ 4096
#define M_  (B_ * T_)   // 32768 rows

typedef __attribute__((ext_vector_type(16))) __bf16        v16bf;
typedef __attribute__((ext_vector_type(16))) unsigned short v16u;
typedef __attribute__((ext_vector_type(8)))  float          v8f;

__device__ __forceinline__ unsigned short f2bf(float f) {
  unsigned int u = __builtin_bit_cast(unsigned int, f);
  u += 0x7FFFu + ((u >> 16) & 1u);            // round-to-nearest-even
  return (unsigned short)(u >> 16);
}

// Low 32 bits of a generic pointer into LDS = workgroup-relative LDS byte
// offset (LDS aperture lives in addr[63:32] on CDNA5).
__device__ __forceinline__ unsigned lds_off32(const void* p) {
  return (unsigned)(unsigned long long)p;
}

// Gather one 16x32 bf16 WMMA fragment from an LDS tile stored with row stride S
// (elements). "row" = M for A fragments, N for column-major-staged B fragments.
// Per-lane layout (ISA 7.12.2, 16-bit A 16x32): lanes 0-15 rows 0-15 hold
// K {0..7,16..23}; lanes 16-31 hold K {8..15,24..31}; VGPR v holds K pair.
__device__ __forceinline__ v16bf frag_ld(const unsigned short* base, int row0, int S, int lane) {
  const int half = lane >> 4;
  const int r = lane & 15;
  const unsigned short* p = base + (row0 + r) * S;
  v16u u;
#pragma unroll
  for (int v = 0; v < 8; ++v) {
    const int k = ((v & 4) << 2) + half * 8 + ((v & 3) << 1);  // (v>=4?16:0)+half*8+(v%4)*2
    const unsigned int pair = *(const unsigned int*)(p + k);   // k even + even stride -> 4B aligned
    u[2 * v]     = (unsigned short)(pair & 0xFFFFu);
    u[2 * v + 1] = (unsigned short)(pair >> 16);
  }
  return __builtin_bit_cast(v16bf, u);
}

// ---------------------------------------------------------------------------
// Kernel 0: one-time fp32 -> bf16 conversion of the two weight matrices.
// ---------------------------------------------------------------------------
__global__ __launch_bounds__(256)
void k_convert(const float* __restrict__ wc, const float* __restrict__ wp,
               unsigned short* __restrict__ wc_bf, unsigned short* __restrict__ wp_bf) {
  const int i = blockIdx.x * 256 + threadIdx.x;
  if (i < D_ * V_)  wc_bf[i] = f2bf(wc[i]);
  if (i < D_ * DO_) wp_bf[i] = f2bf(wp[i]);
}

// ---------------------------------------------------------------------------
// Kernel 1: GEMM1 (x @ W_ctc + b) fused with log-softmax argmax / max-logprob.
// One workgroup = 16 full rows x all 256 vocab columns so the softmax
// reduction never leaves LDS. 8 waves, each owns two 16x16 N-tiles.
// ---------------------------------------------------------------------------
__global__ __launch_bounds__(256)
void k_gemm1_softmax(const float* __restrict__ x,
                     const unsigned short* __restrict__ wc_bf,
                     const float* __restrict__ b_ctc,
                     int* __restrict__ pred, float* __restrict__ maxlp) {
  __shared__ __attribute__((aligned(16))) unsigned short As[16 * 34];    // A tile, padded stride
  __shared__ __attribute__((aligned(16))) unsigned short Bs[256 * 34];   // B panel, col-major [n][k]
  __shared__ float Out[16 * 256];
  __shared__ float pmax[16 * 16];
  __shared__ int   parg[16 * 16];
  __shared__ float psum[16 * 16];
  __shared__ float rowmax[16];

  const int tid  = threadIdx.x;
  const int lane = tid & 31;
  const int w    = tid >> 5;          // wave 0..7
  const int row0 = blockIdx.x * 16;

  v8f acc0 = {}; v8f acc1 = {};

  for (int k0 = 0; k0 < D_; k0 += 32) {
    // Stage A: 16x32 of x, converting fp32 -> bf16 on the fly (x read exactly once).
    {
      const int r  = tid >> 4;
      const int kp = (tid & 15) << 1;
      const float2 xv = *(const float2*)(x + (size_t)(row0 + r) * D_ + k0 + kp);
      As[r * 34 + kp]     = f2bf(xv.x);
      As[r * 34 + kp + 1] = f2bf(xv.y);
    }
    // Stage B: 32x256 panel of W_ctc (bf16) transposed into Bs[n*34 + k].
#pragma unroll
    for (int j = 0; j < 16; ++j) {
      const int i  = j * 256 + tid;       // 4096 uint loads
      const int k  = i >> 7;              // 128 uints per k-row
      const int nu = i & 127;
      const unsigned int pair = *(const unsigned int*)(wc_bf + (size_t)(k0 + k) * V_ + (nu << 1));
      const int n = nu << 1;
      Bs[n * 34 + k]       = (unsigned short)(pair & 0xFFFFu);
      Bs[(n + 1) * 34 + k] = (unsigned short)(pair >> 16);
    }
    __syncthreads();
    const v16bf a  = frag_ld(As, 0, 34, lane);
    const v16bf b0 = frag_ld(Bs, w * 32,      34, lane);
    const v16bf b1 = frag_ld(Bs, w * 32 + 16, 34, lane);
    acc0 = __builtin_amdgcn_wmma_f32_16x16x32_bf16(false, a, false, b0, (short)0, acc0, false, false);
    acc1 = __builtin_amdgcn_wmma_f32_16x16x32_bf16(false, a, false, b1, (short)0, acc1, false, false);
    __syncthreads();
  }

  // Spill logits tile (+bias) to LDS. C/D layout: M = vgpr + 8*(lane>=16), N = lane&15.
  {
    const int half = lane >> 4;
    const int c = lane & 15;
#pragma unroll
    for (int j = 0; j < 8; ++j) {
      const int r = j + 8 * half;
      const int c0 = w * 32 + c;
      const int c1 = w * 32 + 16 + c;
      Out[r * 256 + c0] = acc0[j] + b_ctc[c0];
      Out[r * 256 + c1] = acc1[j] + b_ctc[c1];
    }
  }
  __syncthreads();
  // Pass 1: row max + first-occurrence argmax (16 partials per row).
  {
    const int r = tid >> 4, c = tid & 15;
    float m = -INFINITY; int a = 0;
#pragma unroll
    for (int i = 0; i < 16; ++i) {
      const float v = Out[r * 256 + c * 16 + i];
      if (v > m) { m = v; a = c * 16 + i; }
    }
    pmax[r * 16 + c] = m; parg[r * 16 + c] = a;
  }
  __syncthreads();
  if (tid < 16) {
    float m = -INFINITY; int a = 0;
#pragma unroll
    for (int c = 0; c < 16; ++c) {
      const float v = pmax[tid * 16 + c];
      if (v > m) { m = v; a = parg[tid * 16 + c]; }
    }
    rowmax[tid] = m;
    pred[row0 + tid] = a;
  }
  __syncthreads();
  // Pass 2: sum of exp(logit - rowmax); maxlp = -log(sumexp).
  {
    const int r = tid >> 4, c = tid & 15;
    const float m = rowmax[r];
    float s = 0.f;
#pragma unroll
    for (int i = 0; i < 16; ++i) s += __expf(Out[r * 256 + c * 16 + i] - m);
    psum[r * 16 + c] = s;
  }
  __syncthreads();
  if (tid < 16) {
    float s = 0.f;
#pragma unroll
    for (int c = 0; c < 16; ++c) s += psum[tid * 16 + c];
    maxlp[row0 + tid] = -__logf(s);
  }
}

// ---------------------------------------------------------------------------
// Kernel 2: CTC run-length compression. One workgroup per batch sample.
// Sequential scan (T=2048) on thread 0 for group ids / denominators; then all
// 256 threads stream x (float4) and emit comp in bf16 (WMMA A operand).
// ---------------------------------------------------------------------------
__global__ __launch_bounds__(256)
void k_compress(const float* __restrict__ x,
                const unsigned char* __restrict__ pad,   // numpy bool: 1 byte, True = pad
                const int* __restrict__ pred,
                const float* __restrict__ maxlp,
                unsigned short* __restrict__ comp_bf,
                float* __restrict__ pad_post) {
  __shared__ int   pred_s[T_];
  __shared__ float w_s[T_];
  __shared__ int   gid_s[T_];
  __shared__ float denom_s[T_];
  __shared__ int   s_lens;

  const int b = blockIdx.x;
  const int tid = threadIdx.x;
  const size_t base = (size_t)b * T_;

  for (int t = tid; t < T_; t += 256) {
    const int valid = pad[base + t] ? 0 : 1;
    pred_s[t]  = pred[base + t];
    gid_s[t]   = valid;                       // temporarily holds validity
    w_s[t]     = valid ? maxlp[base + t] : 0.f;
    denom_s[t] = 0.f;
  }
  __syncthreads();
  if (tid == 0) {
    int cnt = 0;
    for (int t = 0; t < T_; ++t) {
      const int valid = gid_s[t];
      const int bnd = valid && (t == 0 || pred_s[t] != pred_s[t - 1]);
      cnt += bnd;
      int g = cnt - 1; if (g < 0) g = 0;
      gid_s[t] = g;
      denom_s[g] += w_s[t];
    }
    s_lens = cnt;
  }
  __syncthreads();
  const int lens = s_lens;
  for (int t = tid; t < T_; t += 256) {       // normalized per-group weights
    const int valid = pad[base + t] ? 0 : 1;
    float dn = denom_s[gid_s[t]];
    if (dn == 0.f) dn = 1.f;
    w_s[t] = valid ? w_s[t] / dn : 0.f;
  }
  __syncthreads();
  {
    const int c0 = tid * 4;                   // 4 feature columns per thread
    float a0 = 0.f, a1 = 0.f, a2 = 0.f, a3 = 0.f;
    int cur = 0;
    for (int t = 0; t < T_; ++t) {
      const int g = gid_s[t];
      if (g != cur) {
        unsigned short* cp = comp_bf + (base + cur) * (size_t)D_ + c0;
        cp[0] = f2bf(a0); cp[1] = f2bf(a1); cp[2] = f2bf(a2); cp[3] = f2bf(a3);
        a0 = a1 = a2 = a3 = 0.f;
        cur = g;
      }
      const float nw = w_s[t];
      if (nw != 0.f) {
        const float4 xv = *(const float4*)(x + (base + t) * (size_t)D_ + c0);
        a0 += nw * xv.x; a1 += nw * xv.y; a2 += nw * xv.z; a3 += nw * xv.w;
      }
    }
    unsigned short* cp = comp_bf + (base + cur) * (size_t)D_ + c0;
    cp[0] = f2bf(a0); cp[1] = f2bf(a1); cp[2] = f2bf(a2); cp[3] = f2bf(a3);
  }
  {                                           // zero comp rows [lens, T)
    unsigned int* cz = (unsigned int*)(comp_bf + (base + lens) * (size_t)D_);
    const int n = (T_ - lens) * (D_ / 2);
    for (int i = tid; i < n; i += 256) cz[i] = 0u;
  }
  for (int t = tid; t < T_; t += 256)
    pad_post[base + t] = (t >= lens) ? 1.f : 0.f;
}

// ---------------------------------------------------------------------------
// Kernel 3: GEMM2 (comp @ W_proj + b_proj) -> y (fp32, into d_out).
// 128x128 macro-tile, BK=32; 8 waves as 2(M) x 4(N), each wave 64x32
// (4x2 WMMA tiles -> 8 v_wmma per K-step). A panel is staged via CDNA5
// async global->LDS copies (ASYNCcnt); B panel is transpose-staged so
// fragment gathers stay dual-issue ds_load_b32. LDS stride padded to 34.
// ---------------------------------------------------------------------------
__global__ __launch_bounds__(256)
void k_gemm2(const unsigned short* __restrict__ comp_bf,
             const unsigned short* __restrict__ wp_bf,
             const float* __restrict__ b_proj,
             float* __restrict__ y) {
  __shared__ __attribute__((aligned(16))) unsigned short As[128 * 34];  // row-major [m][k]
  __shared__ __attribute__((aligned(16))) unsigned short Bs[128 * 34];  // col-major [n][k]
  const int tid  = threadIdx.x;
  const int lane = tid & 31;
  const int w    = tid >> 5;
  const int mw   = w >> 2;                    // 0..1
  const int nw   = w & 3;                     // 0..3
  const int rowBase = blockIdx.x * 128;
  const int colBase = blockIdx.y * 128;

  const v8f zero = {};
  v8f acc[4][2];
#pragma unroll
  for (int i = 0; i < 4; ++i)
#pragma unroll
    for (int j = 0; j < 2; ++j) acc[i][j] = zero;

  for (int k0 = 0; k0 < D_; k0 += 32) {
    // A: 128x32 bf16 panel, async copy global->LDS (16B per lane, 2 per thread).
    // Per-lane LDS address handles the +2-element padding per 32-element row.
#pragma unroll
    for (int j = 0; j < 2; ++j) {
      const int i = j * 256 + tid;            // 512 16-byte chunks
      const int r = i >> 2;                   // row 0..127
      const int q = i & 3;                    // 16B chunk within row
      const unsigned lds = lds_off32(&As[r * 34 + q * 8]);
      const unsigned goff = ((unsigned)(rowBase + r) * (unsigned)D_ + (unsigned)k0 + (unsigned)(q * 8)) * 2u;
      asm volatile("global_load_async_to_lds_b128 %0, %1, %2"
                   :: "v"(lds), "v"(goff), "s"(comp_bf)
                   : "memory");
    }
    // Prefetch next K-panel of W_proj into cache while we transpose this one.
    if (k0 + 32 < D_)
      __builtin_prefetch(wp_bf + (size_t)(k0 + 32 + (tid >> 3)) * DO_ + colBase + ((tid & 7) << 4), 0, 0);
    // B: 32x128 bf16 panel, stored transposed: Bs[n*34 + k].
#pragma unroll
    for (int j = 0; j < 8; ++j) {
      const int i  = j * 256 + tid;           // 2048 uint loads
      const int k  = i >> 6;
      const int nu = i & 63;
      const unsigned int v = *(const unsigned int*)(wp_bf + (size_t)(k0 + k) * DO_ + colBase + (nu << 1));
      const int n = nu << 1;
      Bs[n * 34 + k]       = (unsigned short)(v & 0xFFFFu);
      Bs[(n + 1) * 34 + k] = (unsigned short)(v >> 16);
    }
    asm volatile("s_wait_asynccnt 0x0" ::: "memory");  // A panel landed in LDS
    __syncthreads();
    v16bf af[4], bfr[2];
#pragma unroll
    for (int i = 0; i < 4; ++i) af[i]  = frag_ld(As, mw * 64 + i * 16, 34, lane);
#pragma unroll
    for (int j = 0; j < 2; ++j) bfr[j] = frag_ld(Bs, nw * 32 + j * 16, 34, lane);
#pragma unroll
    for (int i = 0; i < 4; ++i)
#pragma unroll
      for (int j = 0; j < 2; ++j)
        acc[i][j] = __builtin_amdgcn_wmma_f32_16x16x32_bf16(false, af[i], false, bfr[j],
                                                            (short)0, acc[i][j], false, false);
    __syncthreads();
  }

  const int half = lane >> 4;
  const int cl = lane & 15;
#pragma unroll
  for (int i = 0; i < 4; ++i) {
#pragma unroll
    for (int j = 0; j < 2; ++j) {
      const int col = colBase + nw * 32 + j * 16 + cl;
      const float bias = b_proj[col];
#pragma unroll
      for (int v = 0; v < 8; ++v) {
        const int row = rowBase + mw * 64 + i * 16 + v + 8 * half;
        y[(size_t)row * DO_ + col] = acc[i][j][v] + bias;
      }
    }
  }
}

// ---------------------------------------------------------------------------
// Kernel 4: in-place LayerNorm over DO=4096. One workgroup per row.
// ---------------------------------------------------------------------------
__global__ __launch_bounds__(256)
void k_layernorm(float* __restrict__ y, const float* __restrict__ g, const float* __restrict__ bta) {
  __shared__ float s1[256];
  __shared__ float s2[256];
  const int tid = threadIdx.x;
  float* p = y + (size_t)blockIdx.x * DO_;
  float vals[16];
  float sum = 0.f, sq = 0.f;
#pragma unroll
  for (int i = 0; i < 16; ++i) {
    const float v = p[tid + (i << 8)];
    vals[i] = v; sum += v; sq += v * v;
  }
  s1[tid] = sum; s2[tid] = sq;
  __syncthreads();
  for (int s = 128; s > 0; s >>= 1) {
    if (tid < s) { s1[tid] += s1[tid + s]; s2[tid] += s2[tid + s]; }
    __syncthreads();
  }
  const float mu  = s1[0] * (1.f / DO_);
  const float var = s2[0] * (1.f / DO_) - mu * mu;
  const float rs  = rsqrtf(var + 1e-5f);
#pragma unroll
  for (int i = 0; i < 16; ++i) {
    const int c = tid + (i << 8);
    p[c] = (vals[i] - mu) * rs * g[c] + bta[c];
  }
}

// ---------------------------------------------------------------------------
extern "C" void kernel_launch(void* const* d_in, const int* in_sizes, int n_in,
                              void* d_out, int out_size, void* d_ws, size_t ws_size,
                              hipStream_t stream) {
  (void)in_sizes; (void)n_in; (void)out_size; (void)ws_size;
  const float*         x      = (const float*)d_in[0];
  const unsigned char* pad    = (const unsigned char*)d_in[1];  // bool mask
  const float*         W_ctc  = (const float*)d_in[2];
  const float*         b_ctc  = (const float*)d_in[3];
  const float*         W_proj = (const float*)d_in[4];
  const float*         b_proj = (const float*)d_in[5];
  const float*         ln_g   = (const float*)d_in[6];
  const float*         ln_b   = (const float*)d_in[7];

  // Workspace layout (~76 MB total):
  char* ws = (char*)d_ws;
  unsigned short* wc_bf   = (unsigned short*)(ws);                         // 512 KB
  unsigned short* wp_bf   = (unsigned short*)(ws + (1u << 20));            // 8 MB
  int*            pred    = (int*)(ws + (9u << 20));                       // 128 KB
  float*          maxlp   = (float*)(ws + (10u << 20));                    // 128 KB
  unsigned short* comp_bf = (unsigned short*)(ws + (12u << 20));           // 64 MB

  float* y        = (float*)d_out;                 // (B,T,DO) fp32
  float* pad_post = y + (size_t)M_ * DO_;          // (B,T) as 0/1 floats

  k_convert<<<(D_ * DO_ + 255) / 256, 256, 0, stream>>>(W_ctc, W_proj, wc_bf, wp_bf);
  k_gemm1_softmax<<<M_ / 16, 256, 0, stream>>>(x, wc_bf, b_ctc, pred, maxlp);
  k_compress<<<B_, 256, 0, stream>>>(x, pad, pred, maxlp, comp_bf, pad_post);
  dim3 g3(M_ / 128, DO_ / 128);
  k_gemm2<<<g3, 256, 0, stream>>>(comp_bf, wp_bf, b_proj, y);
  k_layernorm<<<M_, 256, 0, stream>>>(y, ln_g, ln_b);
}
